// FEM0222_55439437857349
// MI455X (gfx1250) — compile-verified, matrix operand order checked by reading, not run on hardware
//
#include <hip/hip_runtime.h>

// ---------------- problem constants ----------------
#define BB   8          // batch
#define CIN  256        // input channels
#define CC   128        // bottleneck channels
#define NN   2304       // H*W = 48*48
#define RR   8          // gate hidden (c/16)

typedef __attribute__((ext_vector_type(16))) _Float16 v16h;
typedef __attribute__((ext_vector_type(8)))  float    v8f;
typedef int v4i_vs __attribute__((vector_size(16)));
typedef __attribute__((address_space(1))) v4i_vs* gptr_v4i;
typedef __attribute__((address_space(3))) v4i_vs* lptr_v4i;

#if __has_builtin(__builtin_amdgcn_global_load_async_to_lds_b128)
#define USE_ASYNC_LDS 1
#else
#define USE_ASYNC_LDS 0
#endif

// ---------------- f32 -> f16 convert ----------------
__global__ void cvt_f32_f16(const float* __restrict__ in, _Float16* __restrict__ out, size_t n) {
    size_t i = (size_t)blockIdx.x * blockDim.x + threadIdx.x;
    if (i < n) out[i] = (_Float16)in[i];
}

// ---------------- generic WMMA GEMM ----------------
// C[M,N] = A[M,K] (f16, row-major, lda) x B[K,N] (f16, row-major, ldb)
// 128 threads = 4 waves; block tile 64x64; K-step 32.
// A tile staged with GLOBAL_LOAD_ASYNC_TO_LDS_B128 (ASYNCcnt path) when available.
// EPI: 0 = +bias, BN -> f32[M,N] and f16[M,N]
//      1 = +bias      -> f16 transposed [N, ldt]
//      2 = +bias      -> f16[M,N]
//      3 = plain      -> f16[M,N]
//      4 = plain      -> f32[M,N]
//      5 = BN + ReLU  -> f32[M,N]
#define TM 64
#define TN 64
#define TK 32

template<int EPI>
__global__ __launch_bounds__(128)
void gemm_wmma(const _Float16* __restrict__ A, const _Float16* __restrict__ B,
               float* __restrict__ out32, _Float16* __restrict__ out16,
               const float* __restrict__ bias,
               const float* __restrict__ bng, const float* __restrict__ bnb,
               const float* __restrict__ bnm, const float* __restrict__ bnv,
               int M, int N, int K, int lda, int ldb, int ldo, int ldt,
               size_t sA, size_t sB, size_t sO32, size_t sO16)
{
    __shared__ __align__(16) _Float16 As[TM][TK];   // [m][k]
    __shared__ __align__(16) _Float16 Bs[TN][TK];   // [n][k] (transposed)

    const int bz = blockIdx.z;
    A += (size_t)bz * sA;
    B += (size_t)bz * sB;
    if (out32) out32 += (size_t)bz * sO32;
    if (out16) out16 += (size_t)bz * sO16;

    const int m0   = blockIdx.y * TM;
    const int n0   = blockIdx.x * TN;
    const int tid  = threadIdx.x;
    const int wave = tid >> 5;
    const int lane = tid & 31;
    const int g    = lane >> 4;   // half-wave group
    const int r    = lane & 15;

    v8f acc[4] = {};

    for (int k0 = 0; k0 < K; k0 += TK) {
        // ---- stage A tile: 64x32 halves via async 128-bit copies to LDS ----
        {
            int idx = tid * 8;
            #pragma unroll
            for (int rep = 0; rep < 2; ++rep, idx += 1024) {
                int row = idx >> 5, col = idx & 31;
#if USE_ASYNC_LDS
                __builtin_amdgcn_global_load_async_to_lds_b128(
                    (gptr_v4i)(A + (size_t)(m0 + row) * lda + k0 + col),
                    (lptr_v4i)&As[row][col],
                    0, 0);
#else
                const int4* src = (const int4*)(A + (size_t)(m0 + row) * lda + k0 + col);
                *(int4*)&As[row][col] = *src;
#endif
            }
        }
        // ---- stage B tile: read 32x64 coalesced along N, store transposed [n][k] ----
        {
            int cidx = tid;
            #pragma unroll
            for (int rep = 0; rep < 2; ++rep, cidx += 128) {
                int kk = cidx >> 3;          // 0..31
                int nn = (cidx & 7) * 8;     // 0..56
                int4 v = *(const int4*)(B + (size_t)(k0 + kk) * ldb + n0 + nn);
                const _Float16* hv = (const _Float16*)&v;
                #pragma unroll
                for (int j = 0; j < 8; ++j) Bs[nn + j][kk] = hv[j];
            }
        }
#if USE_ASYNC_LDS
        asm volatile("s_wait_asynccnt 0" ::: "memory");
#endif
        __syncthreads();

        // ---- build A fragment (16x32): lane r holds row M=r; K runs {8g..8g+7, 16+8g..16+8g+7}
        v16h af;
        {
            const _Float16* arow = &As[wave * 16 + r][0];
            #pragma unroll
            for (int h = 0; h < 8; ++h) { af[h] = arow[8 * g + h]; af[h + 8] = arow[16 + 8 * g + h]; }
        }
        // ---- 4 WMMAs across the 64-wide N tile ----
        #pragma unroll
        for (int cb = 0; cb < 4; ++cb) {
            v16h bf;
            const _Float16* brow = &Bs[cb * 16 + r][0];
            #pragma unroll
            for (int h = 0; h < 8; ++h) { bf[h] = brow[8 * g + h]; bf[h + 8] = brow[16 + 8 * g + h]; }
            acc[cb] = __builtin_amdgcn_wmma_f32_16x16x32_f16(false, af, false, bf,
                                                             (short)0, acc[cb], false, false);
        }
        __syncthreads();
    }

    // ---- epilogue: acc[cb][j] is element (M = m0+16*wave+j+8*g, N = n0+16*cb+r) ----
    #pragma unroll
    for (int cb = 0; cb < 4; ++cb) {
        #pragma unroll
        for (int j = 0; j < 8; ++j) {
            int mm = m0 + wave * 16 + j + 8 * g;
            int nn = n0 + cb * 16 + r;
            float y = acc[cb][j];
            if constexpr (EPI == 0) {
                y += bias[mm];
                y = (y - bnm[mm]) * rsqrtf(bnv[mm] + 1e-5f) * bng[mm] + bnb[mm];
                out32[(size_t)mm * ldo + nn] = y;
                out16[(size_t)mm * ldo + nn] = (_Float16)y;
            } else if constexpr (EPI == 1) {
                y += bias[mm];
                out16[(size_t)nn * ldt + mm] = (_Float16)y;
            } else if constexpr (EPI == 2) {
                y += bias[mm];
                out16[(size_t)mm * ldo + nn] = (_Float16)y;
            } else if constexpr (EPI == 3) {
                out16[(size_t)mm * ldo + nn] = (_Float16)y;
            } else if constexpr (EPI == 4) {
                out32[(size_t)mm * ldo + nn] = y;
            } else {
                y = (y - bnm[mm]) * rsqrtf(bnv[mm] + 1e-5f) * bng[mm] + bnb[mm];
                out32[(size_t)mm * ldo + nn] = fmaxf(y, 0.f);
            }
        }
    }
}

// ---------------- block reductions ----------------
__device__ inline float blockReduceMax(float v, float* red) {
    #pragma unroll
    for (int k = 16; k; k >>= 1) v = fmaxf(v, __shfl_xor(v, k, 32));
    int warp = threadIdx.x >> 5, lane = threadIdx.x & 31;
    if (lane == 0) red[warp] = v;
    __syncthreads();
    if (threadIdx.x == 0) {
        float m = red[0];
        for (int j = 1; j < (int)(blockDim.x >> 5); ++j) m = fmaxf(m, red[j]);
        red[0] = m;
    }
    __syncthreads();
    float m = red[0];
    __syncthreads();
    return m;
}

__device__ inline float blockReduceSum(float v, float* red) {
    #pragma unroll
    for (int k = 16; k; k >>= 1) v += __shfl_xor(v, k, 32);
    int warp = threadIdx.x >> 5, lane = threadIdx.x & 31;
    if (lane == 0) red[warp] = v;
    __syncthreads();
    if (threadIdx.x == 0) {
        float m = red[0];
        for (int j = 1; j < (int)(blockDim.x >> 5); ++j) m += red[j];
        red[0] = m;
    }
    __syncthreads();
    float m = red[0];
    __syncthreads();
    return m;
}

// softmax over rows of A[b][row][*] (f16 in, f16 out)
__global__ __launch_bounds__(256)
void softmax_rows(const _Float16* __restrict__ A, _Float16* __restrict__ out,
                  int ncols, size_t bstride)
{
    __shared__ float red[8];
    const _Float16* a = A  + (size_t)blockIdx.y * bstride + (size_t)blockIdx.x * ncols;
    _Float16*       o = out + (size_t)blockIdx.y * bstride + (size_t)blockIdx.x * ncols;
    float mx = -1e30f;
    for (int i = threadIdx.x; i < ncols; i += 256) mx = fmaxf(mx, (float)a[i]);
    mx = blockReduceMax(mx, red);
    float s = 0.f;
    for (int i = threadIdx.x; i < ncols; i += 256) s += __expf((float)a[i] - mx);
    s = blockReduceSum(s, red);
    float inv = 1.f / s;
    for (int i = threadIdx.x; i < ncols; i += 256)
        o[i] = (_Float16)(__expf((float)a[i] - mx) * inv);
}

// att_q[b][m][n] = softmax_n A[b][n][m]
__global__ __launch_bounds__(256)
void softmax_cols(const _Float16* __restrict__ A, _Float16* __restrict__ out,
                  int nrows, int ncols, size_t bstride)
{
    __shared__ float red[8];
    const int m = blockIdx.x;
    const _Float16* a = A  + (size_t)blockIdx.y * bstride;
    _Float16*       o = out + (size_t)blockIdx.y * bstride + (size_t)m * nrows;
    float mx = -1e30f;
    for (int n = threadIdx.x; n < nrows; n += 256)
        mx = fmaxf(mx, (float)a[(size_t)n * ncols + m]);
    mx = blockReduceMax(mx, red);
    float s = 0.f;
    for (int n = threadIdx.x; n < nrows; n += 256)
        s += __expf((float)a[(size_t)n * ncols + m] - mx);
    s = blockReduceSum(s, red);
    float inv = 1.f / s;
    for (int n = threadIdx.x; n < nrows; n += 256)
        o[n] = (_Float16)(__expf((float)a[(size_t)n * ncols + m] - mx) * inv);
}

// global average pool over spatial: t[b][cc][*] -> pool[b*CC+cc]
__global__ __launch_bounds__(256)
void pool_kernel(const float* __restrict__ t, float* __restrict__ pool, int n)
{
    __shared__ float red[8];
    const float* p = t + ((size_t)blockIdx.y * CC + blockIdx.x) * n;
    float s = 0.f;
    for (int i = threadIdx.x; i < n; i += 256) s += p[i];
    s = blockReduceSum(s, red);
    if (threadIdx.x == 0) pool[blockIdx.y * CC + blockIdx.x] = s / (float)n;
}

// gate = sigmoid(relu(pool@w1^T+b1)@w2^T+b2), one block per batch, 128 threads
__global__ __launch_bounds__(128)
void gate_mlp(const float* __restrict__ pool,
              const float* __restrict__ w1, const float* __restrict__ b1,
              const float* __restrict__ w2, const float* __restrict__ b2,
              float* __restrict__ gate)
{
    __shared__ float h[RR];
    const int b = blockIdx.x, t = threadIdx.x;
    const float* p = pool + b * CC;
    if (t < RR) {
        float a = b1[t];
        for (int k = 0; k < CC; ++k) a += p[k] * w1[t * CC + k];
        h[t] = fmaxf(a, 0.f);
    }
    __syncthreads();
    float a = b2[t];
    #pragma unroll
    for (int j = 0; j < RR; ++j) a += h[j] * w2[t * RR + j];
    gate[b * CC + t] = 1.f / (1.f + __expf(-a));
}

// E[b][cc][n] = gate[b][cc] * pbuf[b][n][cc] + t[b][cc][n]  -> f32 out + f16 into cat buffer
__global__ void fuse_E(const float* __restrict__ gate, const float* __restrict__ pbuf,
                       const float* __restrict__ tbuf, float* __restrict__ outE,
                       _Float16* __restrict__ cat16, int chanOff)
{
    size_t i = (size_t)blockIdx.x * blockDim.x + threadIdx.x;
    const size_t total = (size_t)BB * CC * NN;
    if (i >= total) return;
    int b  = (int)(i / ((size_t)CC * NN));
    int rem = (int)(i % ((size_t)CC * NN));
    int cc = rem / NN;
    int n  = rem % NN;
    float e = gate[b * CC + cc] * pbuf[(size_t)b * NN * CC + (size_t)n * CC + cc] + tbuf[i];
    outE[i] = e;
    cat16[((size_t)b * (2 * CC) + chanOff + cc) * NN + n] = (_Float16)e;
}

// im2col for 3x3/pad1 over cat16 [B, 2C, 48, 48] -> col [B, 2C*9, NN]
__global__ void im2col3x3(const _Float16* __restrict__ cat16, _Float16* __restrict__ col)
{
    size_t i = (size_t)blockIdx.x * blockDim.x + threadIdx.x;
    const size_t total = (size_t)BB * (2 * CC * 9) * NN;
    if (i >= total) return;
    int b    = (int)(i / ((size_t)(2 * CC * 9) * NN));
    int rem  = (int)(i % ((size_t)(2 * CC * 9) * NN));
    int kidx = rem / NN;
    int n    = rem % NN;
    int ci = kidx / 9, kk = kidx % 9;
    int y = n / 48 + kk / 3 - 1;
    int x = n % 48 + kk % 3 - 1;
    _Float16 v = (_Float16)0.f;
    if ((unsigned)y < 48u && (unsigned)x < 48u)
        v = cat16[((size_t)b * (2 * CC) + ci) * NN + y * 48 + x];
    col[i] = v;
}

// ---------------- host launcher ----------------
extern "C" void kernel_launch(void* const* d_in, const int* in_sizes, int n_in,
                              void* d_out, int out_size, void* d_ws, size_t ws_size,
                              hipStream_t stream)
{
    const float* q         = (const float*)d_in[0];
    const float* s         = (const float*)d_in[1];
    const float* trans_s_w = (const float*)d_in[2];
    const float* trans_s_b = (const float*)d_in[3];
    const float* bn_s_g    = (const float*)d_in[4];
    const float* bn_s_b    = (const float*)d_in[5];
    const float* bn_s_m    = (const float*)d_in[6];
    const float* bn_s_v    = (const float*)d_in[7];
    const float* trans_q_w = (const float*)d_in[8];
    const float* trans_q_b = (const float*)d_in[9];
    const float* bn_q_g    = (const float*)d_in[10];
    const float* bn_q_b    = (const float*)d_in[11];
    const float* bn_q_m    = (const float*)d_in[12];
    const float* bn_q_v    = (const float*)d_in[13];
    const float* vq_w      = (const float*)d_in[14];
    const float* vq_b      = (const float*)d_in[15];
    const float* vs_w      = (const float*)d_in[16];
    const float* vs_b      = (const float*)d_in[17];
    const float* key_w     = (const float*)d_in[18];
    const float* key_b     = (const float*)d_in[19];
    const float* query_w   = (const float*)d_in[20];
    const float* query_b   = (const float*)d_in[21];
    const float* gate_w1   = (const float*)d_in[22];
    const float* gate_b1   = (const float*)d_in[23];
    const float* gate_w2   = (const float*)d_in[24];
    const float* gate_b2   = (const float*)d_in[25];
    const float* cat_w     = (const float*)d_in[26];
    const float* cat_g     = (const float*)d_in[27];
    const float* cat_b     = (const float*)d_in[28];
    const float* cat_m     = (const float*)d_in[29];
    const float* cat_v     = (const float*)d_in[30];

    float* out = (float*)d_out;
    float* out_cpam = out;                               // [8,128,48,48]
    float* out_Eq   = out + (size_t)BB * CC * NN;        // [8,128,48,48]
    float* out_Es   = out + (size_t)2 * BB * CC * NN;

    // ---- workspace carve-out ----
    char* ws = (char*)d_ws;
    size_t off = 0;
    auto carve = [&](size_t bytes) -> void* {
        void* p = ws + off;
        off = (off + bytes + 255) & ~(size_t)255;
        return p;
    };
    const size_t szIn16  = (size_t)BB * CIN * NN * 2;   // f16 [8,256,2304]
    const size_t szT32   = (size_t)BB * CC  * NN * 4;   // f32 [8,128,2304]
    const size_t szT16   = (size_t)BB * CC  * NN * 2;
    const size_t szScore = (size_t)BB * NN  * NN * 2;   // f16 [8,2304,2304]

    _Float16* q16    = (_Float16*)carve(szIn16);
    _Float16* s16    = (_Float16*)carve(szIn16);
    _Float16* tws16  = (_Float16*)carve((size_t)CC * CIN * 2);
    _Float16* twq16  = (_Float16*)carve((size_t)CC * CIN * 2);
    _Float16* vqw16  = (_Float16*)carve((size_t)CC * CC * 2);
    _Float16* vsw16  = (_Float16*)carve((size_t)CC * CC * 2);
    _Float16* keyw16 = (_Float16*)carve((size_t)CC * CC * 2);
    _Float16* qryw16 = (_Float16*)carve((size_t)CC * CC * 2);
    _Float16* catw16 = (_Float16*)carve((size_t)CC * (2 * CC * 9) * 2);
    float*    s_t    = (float*)   carve(szT32);
    float*    q_t    = (float*)   carve(szT32);
    _Float16* s_t16  = (_Float16*)carve(szT16);
    _Float16* q_t16  = (_Float16*)carve(szT16);
    _Float16* vqT    = (_Float16*)carve(szT16);   // [B, N, c]
    _Float16* vsT    = (_Float16*)carve(szT16);   // [B, N, c]
    _Float16* kxT    = (_Float16*)carve(szT16);   // [B, N, c]
    _Float16* qx16   = (_Float16*)carve(szT16);   // [B, c, N]
    float*    ps32   = (float*)   carve(szT32);   // [B, N, c]
    float*    qs32   = (float*)   carve(szT32);   // [B, N, c]
    float*    pool_s = (float*)   carve((size_t)BB * CC * 4);
    float*    pool_q = (float*)   carve((size_t)BB * CC * 4);
    float*    gate_s = (float*)   carve((size_t)BB * CC * 4);
    float*    gate_q = (float*)   carve((size_t)BB * CC * 4);
    _Float16* cat16  = (_Float16*)carve(szIn16);  // [B, 256, 2304]
    _Float16* att_s  = (_Float16*)carve(szScore);
    _Float16* att_q  = (_Float16*)carve(szScore);
    _Float16* Abuf   = (_Float16*)carve(szScore); // scores; reused as im2col buffer
    _Float16* colbuf = Abuf;
    (void)ws_size; (void)in_sizes; (void)n_in; (void)out_size;

    auto cvt = [&](const float* src, _Float16* dst, size_t n) {
        int grid = (int)((n + 255) / 256);
        cvt_f32_f16<<<grid, 256, 0, stream>>>(src, dst, n);
    };
    cvt(q, q16, (size_t)BB * CIN * NN);
    cvt(s, s16, (size_t)BB * CIN * NN);
    cvt(trans_s_w, tws16, (size_t)CC * CIN);
    cvt(trans_q_w, twq16, (size_t)CC * CIN);
    cvt(vq_w,   vqw16,  (size_t)CC * CC);
    cvt(vs_w,   vsw16,  (size_t)CC * CC);
    cvt(key_w,  keyw16, (size_t)CC * CC);
    cvt(query_w,qryw16, (size_t)CC * CC);
    cvt(cat_w,  catw16, (size_t)CC * 2 * CC * 9);

    const size_t sIn  = (size_t)CIN * NN;
    const size_t sT   = (size_t)CC * NN;
    const size_t sSc  = (size_t)NN * NN;

    // trans_s / trans_q : [128,2304] = W[128,256] x X[256,2304], bias+BN, f32+f16 out
    gemm_wmma<0><<<dim3(NN / TN, CC / TM, BB), 128, 0, stream>>>(
        tws16, s16, s_t, s_t16, trans_s_b, bn_s_g, bn_s_b, bn_s_m, bn_s_v,
        CC, NN, CIN, CIN, NN, NN, 0, 0, sIn, sT, sT);
    gemm_wmma<0><<<dim3(NN / TN, CC / TM, BB), 128, 0, stream>>>(
        twq16, q16, q_t, q_t16, trans_q_b, bn_q_g, bn_q_b, bn_q_m, bn_q_v,
        CC, NN, CIN, CIN, NN, NN, 0, 0, sIn, sT, sT);

    // v_s, v_q, k_x : bias, f16 transposed [N, c]
    gemm_wmma<1><<<dim3(NN / TN, CC / TM, BB), 128, 0, stream>>>(
        vsw16, s_t16, nullptr, vsT, vs_b, nullptr, nullptr, nullptr, nullptr,
        CC, NN, CC, CC, NN, 0, CC, 0, sT, 0, sT);
    gemm_wmma<1><<<dim3(NN / TN, CC / TM, BB), 128, 0, stream>>>(
        vqw16, q_t16, nullptr, vqT, vq_b, nullptr, nullptr, nullptr, nullptr,
        CC, NN, CC, CC, NN, 0, CC, 0, sT, 0, sT);
    gemm_wmma<1><<<dim3(NN / TN, CC / TM, BB), 128, 0, stream>>>(
        keyw16, s_t16, nullptr, kxT, key_b, nullptr, nullptr, nullptr, nullptr,
        CC, NN, CC, CC, NN, 0, CC, 0, sT, 0, sT);
    // q_x : bias, f16 [c, N]
    gemm_wmma<2><<<dim3(NN / TN, CC / TM, BB), 128, 0, stream>>>(
        qryw16, q_t16, nullptr, qx16, query_b, nullptr, nullptr, nullptr, nullptr,
        CC, NN, CC, CC, NN, NN, 0, 0, sT, 0, sT);

    // A_s[b] = kxT[N,c] x qx[c,N]  -> f16 [N,N]
    gemm_wmma<3><<<dim3(NN / TN, NN / TM, BB), 128, 0, stream>>>(
        kxT, qx16, nullptr, Abuf, nullptr, nullptr, nullptr, nullptr, nullptr,
        NN, NN, CC, CC, NN, NN, 0, sT, sT, 0, sSc);

    // softmaxes
    softmax_rows<<<dim3(NN, BB), 256, 0, stream>>>(Abuf, att_s, NN, sSc);
    softmax_cols<<<dim3(NN, BB), 256, 0, stream>>>(Abuf, att_q, NN, NN, sSc);

    // p_s = att_s x v_s ; q_s = att_q x v_q  -> f32 [N, c]
    gemm_wmma<4><<<dim3(CC / TN, NN / TM, BB), 128, 0, stream>>>(
        att_s, vsT, ps32, nullptr, nullptr, nullptr, nullptr, nullptr, nullptr,
        NN, CC, NN, NN, CC, CC, 0, sSc, sT, sT, 0);
    gemm_wmma<4><<<dim3(CC / TN, NN / TM, BB), 128, 0, stream>>>(
        att_q, vqT, qs32, nullptr, nullptr, nullptr, nullptr, nullptr, nullptr,
        NN, CC, NN, NN, CC, CC, 0, sSc, sT, sT, 0);

    // channel gates
    pool_kernel<<<dim3(CC, BB), 256, 0, stream>>>(s_t, pool_s, NN);
    pool_kernel<<<dim3(CC, BB), 256, 0, stream>>>(q_t, pool_q, NN);
    gate_mlp<<<BB, CC, 0, stream>>>(pool_s, gate_w1, gate_b1, gate_w2, gate_b2, gate_s);
    gate_mlp<<<BB, CC, 0, stream>>>(pool_q, gate_w1, gate_b1, gate_w2, gate_b2, gate_q);

    // residual fusion -> E_q, E_s (f32 outputs) + f16 concat buffer
    {
        size_t tot = (size_t)BB * CC * NN;
        int grid = (int)((tot + 255) / 256);
        fuse_E<<<grid, 256, 0, stream>>>(gate_q, qs32, q_t, out_Eq, cat16, 0);
        fuse_E<<<grid, 256, 0, stream>>>(gate_s, ps32, s_t, out_Es, cat16, CC);
    }

    // im2col (reuses score buffer) + final conv GEMM with BN+ReLU
    {
        size_t tot = (size_t)BB * (2 * CC * 9) * NN;
        int grid = (int)((tot + 255) / 256);
        im2col3x3<<<grid, 256, 0, stream>>>(cat16, colbuf);
    }
    gemm_wmma<5><<<dim3(NN / TN, CC / TM, BB), 128, 0, stream>>>(
        catw16, colbuf, out_cpam, nullptr, nullptr, cat_g, cat_b, cat_m, cat_v,
        CC, NN, 2 * CC * 9, 2 * CC * 9, NN, NN, 0, 0, sSc, sT, 0);
}
